// Detector_11227044512403
// MI455X (gfx1250) — matrix operand order repaired and checked by reading, not compiled, not run on hardware
//
#include <hip/hip_runtime.h>
#include <math.h>

#define USE_ASYNC 1

typedef __attribute__((ext_vector_type(16))) _Float16 v16h;
typedef __attribute__((ext_vector_type(8)))  float    v8f;

union AFrag { v16h v; _Float16 e[16]; };
union CFrag { v8f  v; float    e[8];  };

// ---------------------------------------------------------------------------
// Weight pre-pass: w[k][co] (f32, HWIO flattened) -> wT[co][k] (f16).
// Makes B-tile staging a raw contiguous copy (async-to-LDS eligible).
// ---------------------------------------------------------------------------
__global__ void transpose_w_kernel(const float* __restrict__ w,
                                   _Float16* __restrict__ wt,
                                   int Ktot, int Cout)
{
    const int i = blockIdx.x * blockDim.x + threadIdx.x;   // over Cout*Ktot
    if (i >= Ktot * Cout) return;
    const int co = i / Ktot;
    const int k  = i - co * Ktot;
    wt[i] = (_Float16)w[(size_t)k * Cout + co];
}

// ===========================================================================
// Tiled implicit-GEMM conv. f16 activations, f16 pre-transposed weights.
// Requires Cin % 32 == 0, Cout % 128 == 0, (Ho*Wo) % 64 == 0.
// Block = 256 threads (8 waves) -> 64(M) x 128(N) output tile.
// K loop = (kh,kw) x 32-wide ci chunks (contiguous, no div/mod in hot loop).
// Per chunk: A tile (64x32 h) + B tile (128x32 h) staged via
// global_load_async_to_lds_b128; per wave 10x ds_load_b128 -> 4 WMMAs.
// ===========================================================================
__global__ void __launch_bounds__(256)
conv_wmma_tiled(const _Float16* __restrict__ in, const _Float16* __restrict__ wt,
                const float* __restrict__ bias, _Float16* __restrict__ out,
                int H, int W, int Cin, int Ho, int Wo, int Cout,
                int KH, int KW, int stride, int padH, int padW,
                int relu, const _Float16* __restrict__ addsrc, int addW)
{
    __shared__ __align__(16) _Float16 As[64 * 32];    // [m][ci]    4 KB
    __shared__ __align__(16) _Float16 Bt[128 * 32];   // [n][ci]    8 KB

    const int tid   = threadIdx.x;
    const int lane  = tid & 31;
    const int wave  = tid >> 5;
    const int wavem = wave & 3;   // 0..3 -> 16-row M sub-tile
    const int waven = wave >> 2;  // 0..1 -> 64-col N sub-tile
    const int mBase = blockIdx.x * 64;
    const int nBase = blockIdx.y * 128;
    const int Ktot  = KH * KW * Cin;

    // ---- A staging: thread -> (row, 8-half ci chunk) ----
    const int smi = tid >> 2;          // 0..63
    const int sc8 = (tid & 3) * 8;     // 0,8,16,24
    const int sm  = mBase + smi;
    const int sy  = sm / Wo, sx = sm % Wo;

    // ---- B staging: thread -> (col n, 16-half k run = 2 copies) ----
    const int sbn = tid >> 1;          // 0..127
    const int sbk = (tid & 1) * 16;    // 0,16

    // ---- fragment coords (16-bit A 16x32 / B 32x16, contiguous k = 16*(lane/16) + s in A and B) ----
    const int asel = lane >> 4;
    const int arow = wavem * 16 + (lane & 15);
    const int bcol = waven * 64 + (lane & 15);
    const int bkb  = (lane >> 4) * 16;

    CFrag acc0, acc1, acc2, acc3;
#pragma unroll
    for (int i = 0; i < 8; ++i) {
        acc0.e[i] = 0.0f; acc1.e[i] = 0.0f; acc2.e[i] = 0.0f; acc3.e[i] = 0.0f;
    }

    for (int kh = 0; kh < KH; ++kh) {
        const int iy = sy * stride + kh - padH;
        for (int kw = 0; kw < KW; ++kw) {
            const int ix   = sx * stride + kw - padW;
            const bool inb = (iy >= 0 && iy < H && ix >= 0 && ix < W);
            const size_t inBase = ((size_t)iy * W + ix) * Cin;   // valid iff inb
            const int kpix = (kh * KW + kw) * Cin;

            for (int ci0 = 0; ci0 < Cin; ci0 += 32) {
                // ---- stage A: raw 16B copy (async) or zero fill ----
                if (inb) {
#if USE_ASYNC && defined(__HIP_DEVICE_COMPILE__)
                    unsigned lofsA = (unsigned)(uintptr_t)&As[smi * 32 + sc8];
                    unsigned long long gA =
                        (unsigned long long)(uintptr_t)(in + inBase + ci0 + sc8);
                    asm volatile("global_load_async_to_lds_b128 %0, %1, off"
                                 :: "v"(lofsA), "v"(gA) : "memory");
#else
                    *(int4*)&As[smi * 32 + sc8] =
                        *(const int4*)&in[inBase + ci0 + sc8];
#endif
                } else {
                    int4 z; z.x = z.y = z.z = z.w = 0;
                    *(int4*)&As[smi * 32 + sc8] = z;
                }

                // ---- stage B: two raw 16B copies from wT[co][k] ----
                {
                    const _Float16* wp = wt + (size_t)(nBase + sbn) * Ktot
                                            + kpix + ci0 + sbk;
#if USE_ASYNC && defined(__HIP_DEVICE_COMPILE__)
                    unsigned lofsB = (unsigned)(uintptr_t)&Bt[sbn * 32 + sbk];
                    unsigned long long gB0 = (unsigned long long)(uintptr_t)wp;
                    unsigned long long gB1 = (unsigned long long)(uintptr_t)(wp + 8);
                    asm volatile("global_load_async_to_lds_b128 %0, %1, off"
                                 :: "v"(lofsB), "v"(gB0) : "memory");
                    asm volatile("global_load_async_to_lds_b128 %0, %1, off offset:16"
                                 :: "v"(lofsB), "v"(gB1) : "memory");
#else
                    *(int4*)&Bt[sbn * 32 + sbk]     = *(const int4*)wp;
                    *(int4*)&Bt[sbn * 32 + sbk + 8] = *(const int4*)(wp + 8);
#endif
                }

#if USE_ASYNC && defined(__HIP_DEVICE_COMPILE__)
                asm volatile("s_wait_asynccnt 0" ::: "memory");
#endif
                __syncthreads();

                // ---- fragments from LDS: contiguous 16B runs ----
                AFrag a, b0, b1, b2, b3;
                ((int4*)a.e)[0]  = *(const int4*)&As[arow * 32 + asel * 16];
                ((int4*)a.e)[1]  = *(const int4*)&As[arow * 32 + asel * 16 + 8];
                ((int4*)b0.e)[0] = *(const int4*)&Bt[(bcol +  0) * 32 + bkb];
                ((int4*)b0.e)[1] = *(const int4*)&Bt[(bcol +  0) * 32 + bkb + 8];
                ((int4*)b1.e)[0] = *(const int4*)&Bt[(bcol + 16) * 32 + bkb];
                ((int4*)b1.e)[1] = *(const int4*)&Bt[(bcol + 16) * 32 + bkb + 8];
                ((int4*)b2.e)[0] = *(const int4*)&Bt[(bcol + 32) * 32 + bkb];
                ((int4*)b2.e)[1] = *(const int4*)&Bt[(bcol + 32) * 32 + bkb + 8];
                ((int4*)b3.e)[0] = *(const int4*)&Bt[(bcol + 48) * 32 + bkb];
                ((int4*)b3.e)[1] = *(const int4*)&Bt[(bcol + 48) * 32 + bkb + 8];

                acc0.v = __builtin_amdgcn_wmma_f32_16x16x32_f16(
                             false, a.v, false, b0.v, (short)0, acc0.v, false, false);
                acc1.v = __builtin_amdgcn_wmma_f32_16x16x32_f16(
                             false, a.v, false, b1.v, (short)0, acc1.v, false, false);
                acc2.v = __builtin_amdgcn_wmma_f32_16x16x32_f16(
                             false, a.v, false, b2.v, (short)0, acc2.v, false, false);
                acc3.v = __builtin_amdgcn_wmma_f32_16x16x32_f16(
                             false, a.v, false, b3.v, (short)0, acc3.v, false, false);
                __syncthreads();
            }
        }
    }

    // ---- epilogue: bias (+ upsample add) (+ relu), store f16 ----
#pragma unroll
    for (int t = 0; t < 4; ++t) {
        const CFrag& acc = (t == 0) ? acc0 : (t == 1) ? acc1 : (t == 2) ? acc2 : acc3;
        const int n = nBase + waven * 64 + t * 16 + (lane & 15);
        const float bn = bias[n];
#pragma unroll
        for (int v = 0; v < 8; ++v) {
            const int m = mBase + wavem * 16 + v + (lane >> 4) * 8;
            float val = acc.e[v] + bn;
            if (addsrc) {
                const int y = m / Wo, x = m % Wo;
                val += (float)addsrc[((size_t)(y >> 1) * addW + (x >> 1)) * Cout + n];
            }
            if (relu) val = fmaxf(val, 0.0f);
            out[(size_t)m * Cout + n] = (_Float16)val;
        }
    }
}

// ===========================================================================
// Generic fallback conv (conv1: Cin=3,K=147 and obj conv: Cout=9 only).
// ===========================================================================
template <typename TI, typename TO>
__global__ void __launch_bounds__(128)
conv_generic(const TI* __restrict__ in, const float* __restrict__ w,
             const float* __restrict__ bias, TO* __restrict__ out,
             int H, int W, int Cin, int Ho, int Wo, int Cout,
             int KH, int KW, int stride, int padH, int padW, int relu)
{
    const int lane  = threadIdx.x & 31;
    const int wave  = threadIdx.x >> 5;
    const int Mtot  = Ho * Wo;
    const int mBase = (blockIdx.x * 4 + wave) * 16;
    if (mBase >= Mtot) return;
    const int nBase = blockIdx.y * 16;
    const int Ktot  = KH * KW * Cin;

    const int asel = lane >> 4;
    const int am   = mBase + (lane & 15);
    const int ay   = am / Wo, ax = am % Wo;
    const bool aok = (am < Mtot);
    const int bN     = nBase + (lane & 15);
    const int bKbase = (lane >> 4) * 8;

    CFrag acc;
#pragma unroll
    for (int i = 0; i < 8; ++i) acc.e[i] = 0.0f;

    for (int k0 = 0; k0 < Ktot; k0 += 32) {
        AFrag a, b;
#pragma unroll
        for (int h = 0; h < 16; ++h) {
            const int v      = h >> 1;
            const int klocal = ((v < 4) ? (2 * v) : (16 + 2 * (v - 4)))
                               + asel * 8 + (h & 1);
            const int k = k0 + klocal;
            float val = 0.0f;
            if (aok && k < Ktot) {
                const int ci = k % Cin;
                const int r  = k / Cin;
                const int kw = r % KW;
                const int kh = r / KW;
                const int iyy = ay * stride + kh - padH;
                const int ixx = ax * stride + kw - padW;
                if (iyy >= 0 && iyy < H && ixx >= 0 && ixx < W)
                    val = (float)in[((size_t)iyy * W + ixx) * Cin + ci];
            }
            a.e[h] = (_Float16)val;
        }
#pragma unroll
        for (int h = 0; h < 16; ++h) {
            const int k = k0 + bKbase + h + (h & 8);
            float val = (k < Ktot && bN < Cout) ? w[(size_t)k * Cout + bN] : 0.0f;
            b.e[h] = (_Float16)val;
        }
        acc.v = __builtin_amdgcn_wmma_f32_16x16x32_f16(
                    false, a.v, false, b.v, (short)0, acc.v, false, false);
    }

    const int n = nBase + (lane & 15);
#pragma unroll
    for (int v = 0; v < 8; ++v) {
        const int m = mBase + v + (lane >> 4) * 8;
        if (m < Mtot && n < Cout) {
            float val = acc.e[v] + bias[n];
            if (relu) val = fmaxf(val, 0.0f);
            out[(size_t)m * Cout + n] = (TO)val;
        }
    }
}

// ===========================================================================
__global__ void sigmoid_kernel(const float* __restrict__ obj,
                               float* __restrict__ sc, int n)
{
    const int i = blockIdx.x * blockDim.x + threadIdx.x;
    if (i < n) sc[i] = 1.0f / (1.0f + expf(-obj[i]));
}

__global__ void anchors_kernel(float* __restrict__ boxes, int Hh, int Ww, int A)
{
    const int i = blockIdx.x * blockDim.x + threadIdx.x;
    const int n = Hh * Ww * A;
    if (i >= n) return;
    const int rest = i / A;
    const int x = rest % Ww;
    const int y = rest / Ww;
    const float fy = (float)y / (float)(Hh - 1);
    const float fx = (float)x / (float)(Ww - 1);
    boxes[i * 4 + 0] = fminf(fmaxf(fy - 0.05f, 0.0f), 1.0f);
    boxes[i * 4 + 1] = fminf(fmaxf(fx - 0.05f, 0.0f), 1.0f);
    boxes[i * 4 + 2] = fminf(fmaxf(fy + 0.05f, 0.0f), 1.0f);
    boxes[i * 4 + 3] = fminf(fmaxf(fx + 0.05f, 0.0f), 1.0f);
}

__global__ void __launch_bounds__(1024)
nms_kernel(const float* __restrict__ boxes, float* __restrict__ sc,
           float* __restrict__ out, int N, int maxOut)
{
    __shared__ float sv[1024];
    __shared__ int   si[1024];
    __shared__ float sb[5];
    const int tid = threadIdx.x;

    for (int t = 0; t < maxOut; ++t) {
        float bv = -3.4e38f;
        int   bi = 0;
        for (int i = tid; i < N; i += 1024) {
            const float s = sc[i];
            if (s > bv) { bv = s; bi = i; }
        }
        sv[tid] = bv; si[tid] = bi;
        __syncthreads();
        for (int off = 512; off > 0; off >>= 1) {
            if (tid < off) {
                const float ov = sv[tid + off];
                const int   oi = si[tid + off];
                if (ov > sv[tid] || (ov == sv[tid] && oi < si[tid])) {
                    sv[tid] = ov; si[tid] = oi;
                }
            }
            __syncthreads();
        }
        if (tid == 0) {
            const int sel = si[0];
            const float y1 = boxes[sel * 4 + 0], x1 = boxes[sel * 4 + 1];
            const float y2 = boxes[sel * 4 + 2], x2 = boxes[sel * 4 + 3];
            sb[0] = y1; sb[1] = x1; sb[2] = y2; sb[3] = x2;
            sb[4] = (y2 - y1) * (x2 - x1);
            out[t * 4 + 0] = y1; out[t * 4 + 1] = x1;
            out[t * 4 + 2] = y2; out[t * 4 + 3] = x2;
            out[maxOut * 4 + t] = sv[0];
        }
        __syncthreads();
        const float by1 = sb[0], bx1 = sb[1], by2 = sb[2], bx2 = sb[3], ba = sb[4];
        for (int i = tid; i < N; i += 1024) {
            const float y1 = boxes[i * 4 + 0], x1 = boxes[i * 4 + 1];
            const float y2 = boxes[i * 4 + 2], x2 = boxes[i * 4 + 3];
            const float yy1 = fmaxf(by1, y1), xx1 = fmaxf(bx1, x1);
            const float yy2 = fminf(by2, y2), xx2 = fminf(bx2, x2);
            const float inter = fmaxf(yy2 - yy1, 0.0f) * fmaxf(xx2 - xx1, 0.0f);
            const float area  = (y2 - y1) * (x2 - x1);
            const float iou   = inter / (ba + area - inter + 1e-8f);
            if (iou >= 0.5f) sc[i] = -1e30f;
        }
        __syncthreads();
    }
}

// ===========================================================================
static void launch_transpose(const float* w, _Float16* wt, int Ktot, int Cout,
                             hipStream_t stream)
{
    const int n = Ktot * Cout;
    transpose_w_kernel<<<(n + 255) / 256, 256, 0, stream>>>(w, wt, Ktot, Cout);
}

static void launch_tiled(const _Float16* in, const _Float16* wt, const float* b,
                         _Float16* out, int H, int W, int Cin, int Ho, int Wo,
                         int Cout, int KH, int KW, int stride, int padH, int padW,
                         int relu, const _Float16* addsrc, int addW,
                         hipStream_t stream)
{
    dim3 grid((Ho * Wo) / 64, Cout / 128);
    conv_wmma_tiled<<<grid, 256, 0, stream>>>(in, wt, b, out, H, W, Cin, Ho, Wo,
                                              Cout, KH, KW, stride, padH, padW,
                                              relu, addsrc, addW);
}

extern "C" void kernel_launch(void* const* d_in, const int* in_sizes, int n_in,
                              void* d_out, int out_size, void* d_ws, size_t ws_size,
                              hipStream_t stream)
{
    (void)in_sizes; (void)n_in; (void)out_size; (void)ws_size;

    const float* inp = (const float*)d_in[0];   // [2,512,512,3] -> batch 0 only
    const float* w1  = (const float*)d_in[1];   const float* b1  = (const float*)d_in[2];
    const float* w2  = (const float*)d_in[3];   const float* b2  = (const float*)d_in[4];
    const float* w3  = (const float*)d_in[5];   const float* b3  = (const float*)d_in[6];
    const float* w4  = (const float*)d_in[7];   const float* b4  = (const float*)d_in[8];
    const float* w5  = (const float*)d_in[9];   const float* b5  = (const float*)d_in[10];
    const float* lw3 = (const float*)d_in[11];  const float* lb3 = (const float*)d_in[12];
    const float* lw4 = (const float*)d_in[13];  const float* lb4 = (const float*)d_in[14];
    const float* lw5 = (const float*)d_in[15];  const float* lb5 = (const float*)d_in[16];
    const float* sw3 = (const float*)d_in[17];  const float* sb3 = (const float*)d_in[18];
    const float* rw  = (const float*)d_in[23];  const float* rb  = (const float*)d_in[24];
    const float* ow  = (const float*)d_in[25];  const float* ob  = (const float*)d_in[26];

    // ---- workspace carve ----
    char* wsb = (char*)d_ws;
    size_t off = 0;
    auto carveH = [&](size_t n) {
        _Float16* p = (_Float16*)(wsb + off);
        off += ((n * sizeof(_Float16)) + 255) & ~(size_t)255;
        return p;
    };
    auto carveF = [&](size_t n) {
        float* p = (float*)(wsb + off);
        off += ((n * sizeof(float)) + 255) & ~(size_t)255;
        return p;
    };

    // f16 activations
    _Float16* x1   = carveH((size_t)256 * 256 * 64);
    _Float16* x2   = carveH((size_t)128 * 128 * 256);
    _Float16* c3   = carveH((size_t)64 * 64 * 512);
    _Float16* c4   = carveH((size_t)32 * 32 * 1024);
    _Float16* c5   = carveH((size_t)16 * 16 * 2048);
    _Float16* p5   = carveH((size_t)16 * 16 * 256);
    _Float16* p4   = carveH((size_t)32 * 32 * 256);
    _Float16* p3   = carveH((size_t)64 * 64 * 256);
    _Float16* p3s  = carveH((size_t)64 * 64 * 256);
    _Float16* tbuf = carveH((size_t)64 * 64 * 256);
    // f16 transposed weights: wT[co][k], k = (kh*KW+kw)*Cin+ci
    _Float16* w2t  = carveH((size_t)576 * 256);
    _Float16* w3t  = carveH((size_t)2304 * 512);
    _Float16* w4t  = carveH((size_t)4608 * 1024);
    _Float16* w5t  = carveH((size_t)9216 * 2048);
    _Float16* lw5t = carveH((size_t)2048 * 256);
    _Float16* lw4t = carveH((size_t)1024 * 256);
    _Float16* lw3t = carveH((size_t)512 * 256);
    _Float16* sw3t = carveH((size_t)2304 * 256);
    _Float16* rwt  = carveH((size_t)2304 * 256);
    // f32 tail buffers
    float* obj     = carveF((size_t)64 * 64 * 9);
    float* score   = carveF((size_t)64 * 64 * 9);
    float* boxes   = carveF((size_t)64 * 64 * 9 * 4);

    // ---- weight pre-pass: f32 [k][co] -> f16 [co][k] ----
    launch_transpose(w2,  w2t,  576,  256,  stream);
    launch_transpose(w3,  w3t,  2304, 512,  stream);
    launch_transpose(w4,  w4t,  4608, 1024, stream);
    launch_transpose(w5,  w5t,  9216, 2048, stream);
    launch_transpose(lw5, lw5t, 2048, 256,  stream);
    launch_transpose(lw4, lw4t, 1024, 256,  stream);
    launch_transpose(lw3, lw3t, 512,  256,  stream);
    launch_transpose(sw3, sw3t, 2304, 256,  stream);
    launch_transpose(rw,  rwt,  2304, 256,  stream);

    // conv1 (generic: Cin=3): f32 input -> f16 x1. SAME pad 7x7 s2 on 512 -> 2.
    {
        dim3 grid((256 * 256 / 16 + 3) / 4, 64 / 16);
        conv_generic<float, _Float16><<<grid, 128, 0, stream>>>(
            inp, w1, b1, x1, 512, 512, 3, 256, 256, 64, 7, 7, 2, 2, 2, 1);
    }

    // Backbone (tiled, ReLU). 3x3 s2 on even dims -> pad 0.
    launch_tiled(x1, w2t, b2, x2, 256, 256, 64,  128, 128, 256,  3, 3, 2, 0, 0, 1, nullptr, 0, stream);
    launch_tiled(x2, w3t, b3, c3, 128, 128, 256, 64,  64,  512,  3, 3, 2, 0, 0, 1, nullptr, 0, stream);
    launch_tiled(c3, w4t, b4, c4, 64,  64,  512, 32,  32,  1024, 3, 3, 2, 0, 0, 1, nullptr, 0, stream);
    launch_tiled(c4, w5t, b5, c5, 32,  32, 1024, 16,  16,  2048, 3, 3, 2, 0, 0, 1, nullptr, 0, stream);

    // FPN: 1x1 laterals with fused 2x-nearest upsample add, 3x3 smooth.
    launch_tiled(c5, lw5t, lb5, p5, 16, 16, 2048, 16, 16, 256, 1, 1, 1, 0, 0, 0, nullptr, 0, stream);
    launch_tiled(c4, lw4t, lb4, p4, 32, 32, 1024, 32, 32, 256, 1, 1, 1, 0, 0, 0, p5, 16, stream);
    launch_tiled(c3, lw3t, lb3, p3, 64, 64, 512,  64, 64, 256, 1, 1, 1, 0, 0, 0, p4, 32, stream);
    launch_tiled(p3, sw3t, sb3, p3s, 64, 64, 256, 64, 64, 256, 3, 3, 1, 1, 1, 0, nullptr, 0, stream);

    // RPN head: 3x3 ReLU conv (tiled), then 1x1 obj conv (generic, Cout=9).
    launch_tiled(p3s, rwt, rb, tbuf, 64, 64, 256, 64, 64, 256, 3, 3, 1, 1, 1, 1, nullptr, 0, stream);
    {
        dim3 grid((64 * 64 / 16 + 3) / 4, 1);
        conv_generic<_Float16, float><<<grid, 128, 0, stream>>>(
            tbuf, ow, ob, obj, 64, 64, 256, 64, 64, 9, 1, 1, 1, 0, 0, 0);
    }

    // Scores + anchors + greedy NMS tail.
    const int N = 64 * 64 * 9;
    sigmoid_kernel<<<(N + 255) / 256, 256, 0, stream>>>(obj, score, N);
    anchors_kernel<<<(N + 255) / 256, 256, 0, stream>>>(boxes, 64, 64, 9);
    nms_kernel<<<1, 1024, 0, stream>>>(boxes, score, (float*)d_out, N, 100);
}
